// Decoder_37589553775262
// MI455X (gfx1250) — matrix operand order repaired and checked by reading, not compiled, not run on hardware
//
#include <hip/hip_runtime.h>
#include <stdint.h>

// ============================================================================
// MI455X (gfx1250, CDNA5, wave32) decoder, v2.
//   * One workgroup (256 thr = 8 wave32) owns a 32-row batch tile (2 M-tiles)
//     and runs the entire T=20 recurrence on-chip.  Doubling rows/WG halves
//     the per-step L2 weight re-read traffic (each B-fragment feeds 2 WMMAs).
//   * All GEMMs via v_wmma_f32_16x16x32_bf16 (bf16 in, fp32 accumulate).
//   * Dynamic LDS (~249 KB < 320 KB/WG per CDNA5 ISA): x, xbase, 4x h (bf16),
//     o (bf16), 4x c (fp32, wave-exclusive ownership), head scratch, cond.
//   * Prep kernel: weights -> bf16 transposed [N][K] (B-fragment = 2 b128
//     contiguous loads/lane), packed head weights, tvec[t]=ob+tril-sum(oW_ts).
// Input index mapping assumes setup_inputs() dict insertion order, recursively:
//   0 conditions, 1 state_h, 2 state_c, 3 inW, 4 inb,
//   m  @5 : W U b oW ob aW ab mulW mulb slW slb mlatW mlatb slatW slatb rW rb
//   y  @22: W U b oW ob aW ab mulW mulb slW slb
//   f  @33: (same 11), fadj @44: (same 11)
// ============================================================================

#define BB     8192
#define TT     20
#define UU     256
#define CC     5
#define TBROWS 32
#define MT     2      // M-tiles of 16 rows
#define XSTR   264    // bf16 elements per LDS row (528B, 16B aligned)

// dynamic LDS layout (bytes)
#define OFF_XBUF   0
#define OFF_XBBUF  (OFF_XBUF  + TBROWS * XSTR * 2)          // 16896
#define OFF_HBUF   (OFF_XBBUF + TBROWS * XSTR * 2)          // 33792
#define OFF_OBUF   (OFF_HBUF  + 4 * TBROWS * XSTR * 2)      // 101376
#define OFF_CBUF   (OFF_OBUF  + TBROWS * XSTR * 2)          // 118272
#define OFF_HRAW   (OFF_CBUF  + 4 * TBROWS * UU * 4)        // 249344
#define OFF_COND   (OFF_HRAW  + TBROWS * 32 * 4)            // 253440
#define SMEM_BYTES (OFF_COND  + TBROWS * 8 * 4)             // 254464

typedef __attribute__((ext_vector_type(16))) __bf16 v16bf;
typedef __attribute__((ext_vector_type(8)))  float  v8f;

union FragU { uint4 q[2]; v16bf v; };

__device__ __forceinline__ unsigned short f2bf(float f) {
  unsigned int u = __float_as_uint(f);
  unsigned int r = u + 0x7FFFu + ((u >> 16) & 1u);   // round-to-nearest-even
  return (unsigned short)(r >> 16);
}
__device__ __forceinline__ float bf2f(unsigned short h) {
  return __uint_as_float(((unsigned int)h) << 16);
}
__device__ __forceinline__ float sigmoidf_(float x) { return 1.0f / (1.0f + __expf(-x)); }

__device__ __forceinline__ v8f splat8(float x) {
  v8f r;
#pragma unroll
  for (int i = 0; i < 8; ++i) r[i] = x;
  return r;
}

// A-fragment (16x32 bf16, M x K) from row-major bf16 LDS tile.
__device__ __forceinline__ v16bf loadA_lds(const unsigned short* buf, int lane,
                                           int k0, int strideElems) {
  const int m  = lane & 15;
  const int hk = lane >> 4;
  const uint4* p = (const uint4*)(buf + m * strideElems + k0 + hk * 8);
  FragU f;
  f.q[0] = p[0];   // K  k0+hk*8 .. +7
  f.q[1] = p[2];   // K  k0+hk*8+16 .. +23
  return f.v;
}

// B-fragment (32x16 bf16, K x N) from global bf16 weight stored [N][Ktot].
__device__ __forceinline__ v16bf loadB_glb(const unsigned short* Wt, int lane,
                                           int n0, int k0, int Ktot) {
  const int n  = n0 + (lane & 15);
  const int kb = k0 + (lane >> 4) * 8;
  const uint4* p = (const uint4*)(Wt + (size_t)n * Ktot + kb);
  FragU f;
  f.q[0] = p[0];
  f.q[1] = p[2];
  return f.v;
}

__device__ __forceinline__ v8f wmma_bf16(v16bf a, v16bf b, v8f c) {
  return __builtin_amdgcn_wmma_f32_16x16x32_bf16(false, a, false, b,
                                                 (short)0, c, false, false);
}

// ---------------------------------------------------------------------------
// Parameter pointer tables (branch order: 0=m, 1=y, 2=f, 3=fadj)
// ---------------------------------------------------------------------------
struct ParamPtrs {
  const float* W[4];    const float* Uw[4];   const float* gb[4];
  const float* oW[4];   const float* ob[4];
  const float* aW[4];   const float* ab[4];
  const float* mulW[4]; const float* mulb[4];
  const float* slW[4];  const float* slb[4];
  const float* mlatW;   const float* mlatb;
  const float* slatW;   const float* slatb;
  const float* rW;      const float* rb;
  const float* inW;     const float* inb;
};

struct WsPtrs {
  unsigned short* WzT;     // [4][1024][512] bf16 : [W;U] transposed
  unsigned short* oWT;     // [4][256][256]  bf16 : oW[:256] transposed
  unsigned short* headWT;  // [4][32][256]   bf16 : packed head weights, transposed
  unsigned short* inWT;    // [256][256]     bf16 : inW[:256] transposed
  float* headbias;         // [4][32]
  float* tvec;             // [4][20][256]
};

__device__ __forceinline__ const float* selHeadW(const ParamPtrs& P, int b, int sec) {
  switch (sec) {
    case 0: return P.aW[b];
    case 1: return P.mulW[b];
    case 2: return P.slW[b];
    case 3: return b == 0 ? P.mlatW : nullptr;
    case 4: return b == 0 ? P.slatW : nullptr;
    case 5: return b == 0 ? P.rW    : nullptr;
    default: return nullptr;
  }
}
__device__ __forceinline__ const float* selHeadB(const ParamPtrs& P, int b, int sec) {
  switch (sec) {
    case 0: return P.ab[b];
    case 1: return P.mulb[b];
    case 2: return P.slb[b];
    case 3: return b == 0 ? P.mlatb : nullptr;
    case 4: return b == 0 ? P.slatb : nullptr;
    case 5: return b == 0 ? P.rb    : nullptr;
    default: return nullptr;
  }
}

// ---------------------------------------------------------------------------
// Prep kernel: bf16 weight transposes + head packing + tvec
// ---------------------------------------------------------------------------
__global__ __launch_bounds__(256) void prep_kernel(ParamPtrs P, WsPtrs W) {
  const long N1 = 4L * 1024 * 512;   // WzT
  const long N2 = 4L * 256 * 256;    // oWT
  const long N3 = 4L * 32 * 256;     // headWT
  const long N4 = 256L * 256;        // inWT
  const long N5 = 4L * 32;           // headbias
  const long N6 = 4L * TT * 256;     // tvec
  const long TOT = N1 + N2 + N3 + N4 + N5 + N6;

  for (long i = (long)blockIdx.x * blockDim.x + threadIdx.x; i < TOT;
       i += (long)gridDim.x * blockDim.x) {
    long r = i;
    if (r < N1) {                                 // WzT[b][n][k], k<256 -> W, else U
      int b = (int)(r / (1024L * 512)); long t2 = r % (1024L * 512);
      int n = (int)(t2 / 512), k = (int)(t2 % 512);
      float v = (k < 256) ? P.W[b][(long)k * 1024 + n]
                          : P.Uw[b][(long)(k - 256) * 1024 + n];
      W.WzT[r] = f2bf(v);
      continue;
    }
    r -= N1;
    if (r < N2) {                                 // oWT[b][n][k] = oW[k][n]
      int b = (int)(r / (256L * 256)); long t2 = r % (256L * 256);
      int n = (int)(t2 / 256), k = (int)(t2 % 256);
      W.oWT[r] = f2bf(P.oW[b][(long)k * 256 + n]);
      continue;
    }
    r -= N2;
    if (r < N3) {                                 // headWT[b][j][k]
      int b = (int)(r / (32L * 256)); long t2 = r % (32L * 256);
      int j = (int)(t2 / 256), k = (int)(t2 % 256);
      const float* p = selHeadW(P, b, j / 5);
      W.headWT[r] = f2bf(p ? p[k * 5 + (j % 5)] : 0.0f);
      continue;
    }
    r -= N3;
    if (r < N4) {                                 // inWT[n][k] = inW[k][n]
      int n = (int)(r / 256), k = (int)(r % 256);
      W.inWT[r] = f2bf(P.inW[(long)k * 256 + n]);
      continue;
    }
    r -= N4;
    if (r < N5) {                                 // headbias[b][j]
      int b = (int)(r / 32), j = (int)(r % 32);
      const float* p = selHeadB(P, b, j / 5);
      W.headbias[r] = p ? p[j % 5] : 0.0f;
      continue;
    }
    r -= N5;
    {                                             // tvec[b][t][d] = ob + tril-sum
      int b = (int)(r / (TT * 256L)); long t2 = r % (TT * 256L);
      int t = (int)(t2 / 256), d = (int)(t2 % 256);
      float s = P.ob[b][d];
      for (int j = 0; j <= t; ++j) s += P.oW[b][(long)(256 + j) * 256 + d];
      W.tvec[r] = s;
    }
  }
}

// ---------------------------------------------------------------------------
// Main kernel: one WG = 32 batch rows (2 M-tiles), full T-loop on-chip
// ---------------------------------------------------------------------------
struct MainArgs {
  const float* conditions; const float* state_h; const float* state_c;
  const float* gateB[4];          // per-branch LSTM bias [1024]
  const unsigned short* WzT; const unsigned short* oWT;
  const unsigned short* headWT; const unsigned short* inWT;
  const float* headbias; const float* tvec;
  const float* condW;             // inW rows 256..260 : [5][256]
  const float* inb;               // [256]
  float* out;
};

__global__ __launch_bounds__(256) void decoder_main(MainArgs A) {
  extern __shared__ __align__(16) char smem[];
  unsigned short* xbuf  = (unsigned short*)(smem + OFF_XBUF);   // x     [32][XSTR]
  unsigned short* xbbuf = (unsigned short*)(smem + OFF_XBBUF);  // xbase [32][XSTR]
  unsigned short* hbufA = (unsigned short*)(smem + OFF_HBUF);   // h  [4][32][XSTR]
  unsigned short* obuf  = (unsigned short*)(smem + OFF_OBUF);   // o     [32][XSTR]
  float*          cbuf  = (float*)(smem + OFF_CBUF);            // c  [4][32][256]
  float*          hraw  = (float*)(smem + OFF_HRAW);            // head  [32][32]
  float*          condL = (float*)(smem + OFF_COND);            // cond  [32][8]

  const int tid   = threadIdx.x;
  const int wave  = tid >> 5;
  const int lane  = tid & 31;
  const int ucol  = lane & 15;
  const int mhalf = lane >> 4;
  const int row0  = blockIdx.x * TBROWS;

  // ---------- init: cond0, h, c, per-thread condW column ----------
  if (tid < TBROWS * CC) {
    int r = tid / CC, j = tid % CC;
    condL[r * 8 + j] = A.conditions[(size_t)(row0 + r) * (TT * CC) + j];
  }
#pragma unroll 4
  for (int r = 0; r < TBROWS; ++r) {
    unsigned short hv = f2bf(A.state_h[(size_t)(row0 + r) * UU + tid]);
    float          cv = A.state_c[(size_t)(row0 + r) * UU + tid];
#pragma unroll
    for (int b = 0; b < 4; ++b) {
      hbufA[b * TBROWS * XSTR + r * XSTR + tid] = hv;
      cbuf [b * TBROWS * UU   + r * UU   + tid] = cv;
    }
  }
  float cw[5];
#pragma unroll
  for (int j = 0; j < 5; ++j) cw[j] = A.condW[j * 256 + tid];
  __syncthreads();

  // ---------- xbase = enc_h @ inWT + inb (once; enc_h loop-invariant) ------
  {
    v8f acc[2][MT];
#pragma unroll
    for (int n2 = 0; n2 < 2; ++n2) {
      const int n = (wave * 2 + n2) * 16 + ucol;
      acc[n2][0] = splat8(A.inb[n]);
      acc[n2][1] = acc[n2][0];
    }
    for (int ks = 0; ks < 8; ++ks) {
      v16bf a0 = loadA_lds(hbufA,                  lane, ks * 32, XSTR);
      v16bf a1 = loadA_lds(hbufA + 16 * XSTR,      lane, ks * 32, XSTR);
#pragma unroll
      for (int n2 = 0; n2 < 2; ++n2) {
        v16bf bb = loadB_glb(A.inWT, lane, (wave * 2 + n2) * 16, ks * 32, 256);
        acc[n2][0] = wmma_bf16(a0, bb, acc[n2][0]);
        acc[n2][1] = wmma_bf16(a1, bb, acc[n2][1]);
      }
    }
#pragma unroll
    for (int n2 = 0; n2 < 2; ++n2) {
      const int n = (wave * 2 + n2) * 16 + ucol;
#pragma unroll
      for (int mt = 0; mt < MT; ++mt)
#pragma unroll
        for (int v = 0; v < 8; ++v) {
          int m = mt * 16 + v + 8 * mhalf;
          xbbuf[m * XSTR + n] = f2bf(acc[n2][mt][v]);
        }
    }
  }
  __syncthreads();

  // =========================== time loop ===================================
  for (int t = 0; t < TT; ++t) {
    // ---- stage 1: x = xbase + cond @ condW (rank-5 fp32 update) ----
#pragma unroll 4
    for (int r = 0; r < TBROWS; ++r) {
      float s = bf2f(xbbuf[r * XSTR + tid]);
#pragma unroll
      for (int j = 0; j < 5; ++j) s += condL[r * 8 + j] * cw[j];
      xbuf[r * XSTR + tid] = f2bf(s);
    }
    __syncthreads();

#pragma unroll
    for (int b = 0; b < 4; ++b) {
      const unsigned short* Wz  = A.WzT    + (size_t)b * 1024 * 512;
      const unsigned short* oWt = A.oWT    + (size_t)b * 256 * 256;
      const unsigned short* hWt = A.headWT + (size_t)b * 32 * 256;
      unsigned short*       hb  = hbufA    + (size_t)b * TBROWS * XSTR;
      float*                cb  = cbuf     + (size_t)b * TBROWS * UU;
      __builtin_prefetch(oWt, 0, 1);   // warm next GEMM's weight stream in L2

      // ---- z = [x|h] @ [W;U] + bias : wave owns 2 u-tiles x 4 gates x 2 M --
      float hn[2][MT][8];
#pragma unroll
      for (int u2 = 0; u2 < 2; ++u2) {
        const int ut = wave * 2 + u2;
        const int u  = ut * 16 + ucol;
        v8f acc[4][MT];
#pragma unroll
        for (int g = 0; g < 4; ++g) {
          acc[g][0] = splat8(A.gateB[b][g * 256 + u]);
          acc[g][1] = acc[g][0];
        }
        for (int ks = 0; ks < 16; ++ks) {
          const unsigned short* asrc = (ks < 8) ? xbuf : hb;
          const int ak = (ks & 7) * 32;
          v16bf a0 = loadA_lds(asrc,             lane, ak, XSTR);
          v16bf a1 = loadA_lds(asrc + 16 * XSTR, lane, ak, XSTR);
#pragma unroll
          for (int g = 0; g < 4; ++g) {
            v16bf bb = loadB_glb(Wz, lane, g * 256 + ut * 16, ks * 32, 512);
            acc[g][0] = wmma_bf16(a0, bb, acc[g][0]);
            acc[g][1] = wmma_bf16(a1, bb, acc[g][1]);
          }
        }
        // gates + cell update (c rows/units are wave-exclusive)
#pragma unroll
        for (int mt = 0; mt < MT; ++mt)
#pragma unroll
          for (int v = 0; v < 8; ++v) {
            int m = mt * 16 + v + 8 * mhalf;
            float ii = sigmoidf_(acc[0][mt][v]);
            float ff = sigmoidf_(acc[1][mt][v]);
            float gg = tanhf(acc[2][mt][v]);
            float oo = sigmoidf_(acc[3][mt][v]);
            float cn = ff * cb[m * UU + u] + ii * gg;
            cb[m * UU + u] = cn;
            hn[u2][mt][v] = oo * tanhf(cn);
          }
      }
      __syncthreads();   // all waves done reading old h
#pragma unroll
      for (int u2 = 0; u2 < 2; ++u2) {
        const int u = (wave * 2 + u2) * 16 + ucol;
#pragma unroll
        for (int mt = 0; mt < MT; ++mt)
#pragma unroll
          for (int v = 0; v < 8; ++v) {
            int m = mt * 16 + v + 8 * mhalf;
            hb[m * XSTR + u] = f2bf(hn[u2][mt][v]);
          }
      }
      __syncthreads();   // new h visible

      // ---- o = h @ oW[:256] + tvec[b][t] ----
      {
        v8f acc[2][MT];
#pragma unroll
        for (int n2 = 0; n2 < 2; ++n2) {
          const int n = (wave * 2 + n2) * 16 + ucol;
          acc[n2][0] = splat8(A.tvec[((size_t)b * TT + t) * 256 + n]);
          acc[n2][1] = acc[n2][0];
        }
        for (int ks = 0; ks < 8; ++ks) {
          v16bf a0 = loadA_lds(hb,             lane, ks * 32, XSTR);
          v16bf a1 = loadA_lds(hb + 16 * XSTR, lane, ks * 32, XSTR);
#pragma unroll
          for (int n2 = 0; n2 < 2; ++n2) {
            v16bf bb = loadB_glb(oWt, lane, (wave * 2 + n2) * 16, ks * 32, 256);
            acc[n2][0] = wmma_bf16(a0, bb, acc[n2][0]);
            acc[n2][1] = wmma_bf16(a1, bb, acc[n2][1]);
          }
        }
#pragma unroll
        for (int n2 = 0; n2 < 2; ++n2) {
          const int n = (wave * 2 + n2) * 16 + ucol;
#pragma unroll
          for (int mt = 0; mt < MT; ++mt)
#pragma unroll
            for (int v = 0; v < 8; ++v) {
              int m = mt * 16 + v + 8 * mhalf;
              obuf[m * XSTR + n] = f2bf(acc[n2][mt][v]);
            }
        }
      }
      __syncthreads();

      // ---- heads: o @ packedHeadW (N=32) on waves 0..3 (nt=wave&1, mt=wave>>1)
      if (wave < 4) {
        const int nt = wave & 1;
        const int mt = wave >> 1;
        const int j  = nt * 16 + ucol;
        v8f acc = splat8(0.0f);
        for (int ks = 0; ks < 8; ++ks) {
          v16bf a  = loadA_lds(obuf + mt * 16 * XSTR, lane, ks * 32, XSTR);
          v16bf bb = loadB_glb(hWt, lane, nt * 16, ks * 32, 256);
          acc = wmma_bf16(a, bb, acc);
        }
        float hbias = A.headbias[b * 32 + j];
#pragma unroll
        for (int v = 0; v < 8; ++v) {
          int m = mt * 16 + v + 8 * mhalf;
          hraw[m * 32 + j] = acc[v] + hbias;
        }
      }
      __syncthreads();

      // ---- head elementwise + output writes + next cond ----
      if (tid < TBROWS) {
        const int m = tid;
        const size_t row = (size_t)row0 + m;
        const float* hr = hraw + m * 32;
        float mx = hr[0];
#pragma unroll
        for (int j = 1; j < 5; ++j) mx = fmaxf(mx, hr[j]);
        float e[5], s = 0.0f;
#pragma unroll
        for (int j = 0; j < 5; ++j) { e[j] = __expf(hr[j] - mx); s += e[j]; }
        const float inv = 1.0f / s;
        float a5[5], mu[5], s1 = 0.0f;
#pragma unroll
        for (int j = 0; j < 5; ++j) {
          a5[j] = e[j] * inv;
          mu[j] = hr[5 + j];
          s1 += a5[j] * mu[j];
        }
        if (b == 0) {                         // merge branch -> pv width 30
          float* pm = A.out + (row * TT + t) * 30;
          float s2 = 0.0f;
#pragma unroll
          for (int j = 0; j < 5; ++j) {
            float sl   = __expf(hr[10 + j]);
            float mlat = hr[15 + j];
            float slat = __expf(hr[20 + j]);
            float rr   = tanhf(hr[25 + j]);
            pm[j]      = a5[j];
            pm[5 + j]  = mu[j];
            pm[10 + j] = sl;
            pm[15 + j] = mlat;
            pm[20 + j] = slat;
            pm[25 + j] = rr;
            s2 += a5[j] * mlat;
          }
          condL[m * 8 + 0] = s1;
          condL[m * 8 + 1] = s2;
        } else {                              // other branches -> pv width 15
          size_t base = (size_t)BB * TT * 30 + (size_t)(b - 1) * BB * TT * 15;
          float* pv = A.out + base + (row * TT + t) * 15;
#pragma unroll
          for (int j = 0; j < 5; ++j) {
            pv[j]      = a5[j];
            pv[5 + j]  = mu[j];
            pv[10 + j] = __expf(hr[10 + j]);
          }
          condL[m * 8 + 1 + b] = s1;          // b=1->cond[2], b=2->[3], b=3->[4]
        }
      }
      __syncthreads();
    } // branch loop
  } // t loop
}

// ---------------------------------------------------------------------------
// Host-side launcher
// ---------------------------------------------------------------------------
extern "C" void kernel_launch(void* const* d_in, const int* in_sizes, int n_in,
                              void* d_out, int out_size, void* d_ws, size_t ws_size,
                              hipStream_t stream) {
  (void)in_sizes; (void)n_in; (void)out_size; (void)ws_size;
  auto F = [&](int i) { return (const float*)d_in[i]; };

  ParamPtrs P{};
  // Input index map: dict insertion order, recursive (see header comment).
  P.inW = F(3); P.inb = F(4);
  // branch 0 = 'm' (merge) at base 5
  {
    int o = 5;
    P.W[0] = F(o + 0);  P.Uw[0] = F(o + 1);  P.gb[0] = F(o + 2);
    P.oW[0] = F(o + 3); P.ob[0] = F(o + 4);
    P.aW[0] = F(o + 5); P.ab[0] = F(o + 6);
    P.mulW[0] = F(o + 7); P.mulb[0] = F(o + 8);
    P.slW[0] = F(o + 9);  P.slb[0] = F(o + 10);
    P.mlatW = F(o + 11); P.mlatb = F(o + 12);
    P.slatW = F(o + 13); P.slatb = F(o + 14);
    P.rW = F(o + 15);    P.rb = F(o + 16);
  }
  // branches 1..3 = 'y','f','fadj' at bases 22, 33, 44 (11 leaves each)
  const int bases[3] = {22, 33, 44};
  for (int i = 0; i < 3; ++i) {
    int o = bases[i], b = i + 1;
    P.W[b] = F(o + 0);  P.Uw[b] = F(o + 1);  P.gb[b] = F(o + 2);
    P.oW[b] = F(o + 3); P.ob[b] = F(o + 4);
    P.aW[b] = F(o + 5); P.ab[b] = F(o + 6);
    P.mulW[b] = F(o + 7); P.mulb[b] = F(o + 8);
    P.slW[b] = F(o + 9);  P.slb[b] = F(o + 10);
  }

  // Workspace layout (bytes), 16B-aligned segments
  char* w = (char*)d_ws;
  WsPtrs W{};
  size_t off = 0;
  W.WzT      = (unsigned short*)(w + off); off += 4ULL * 1024 * 512 * 2;  // 4 MB
  W.oWT      = (unsigned short*)(w + off); off += 4ULL * 256 * 256 * 2;   // 512 KB
  W.headWT   = (unsigned short*)(w + off); off += 4ULL * 32 * 256 * 2;    // 64 KB
  W.inWT     = (unsigned short*)(w + off); off += 256ULL * 256 * 2;       // 128 KB
  W.headbias = (float*)(w + off);          off += 4ULL * 32 * 4;
  W.tvec     = (float*)(w + off);          off += 4ULL * TT * 256 * 4;

  prep_kernel<<<2048, 256, 0, stream>>>(P, W);

  MainArgs M{};
  M.conditions = F(0); M.state_h = F(1); M.state_c = F(2);
  for (int b = 0; b < 4; ++b) M.gateB[b] = P.gb[b];
  M.WzT = W.WzT; M.oWT = W.oWT; M.headWT = W.headWT; M.inWT = W.inWT;
  M.headbias = W.headbias; M.tvec = W.tvec;
  M.condW = P.inW + 256 * 256;   // rows 256..260 of inW
  M.inb = P.inb;
  M.out = (float*)d_out;

  decoder_main<<<BB / TBROWS, 256, SMEM_BYTES, stream>>>(M);
}
